// CaTSim_33732673143503
// MI455X (gfx1250) — matrix operand order repaired and checked by reading, not compiled, not run on hardware
//
#include <hip/hip_runtime.h>

#define N_NODES 10000
#define N_EDGES 320000
#define F_IN    128
#define F1C     128
#define F2C     64
#define F3C     32
#define KT      16
#define BINS    16
#define MT      625   // N_NODES / 16 (exact)
#define TNW     8     // n-tiles per wave in similarity kernels
#define NG      ((MT + TNW - 1) / TNW)   // 79 n-tile groups

typedef __attribute__((ext_vector_type(2)))  float    v2f;
typedef __attribute__((ext_vector_type(8)))  float    v8f;
typedef __attribute__((ext_vector_type(8)))  _Float16 v8h;
typedef __attribute__((ext_vector_type(16))) _Float16 v16h;

// ---------- utility ----------
__global__ void k_fill(float* __restrict__ p, float v, long n) {
  long i = (long)blockIdx.x * blockDim.x + threadIdx.x;
  if (i < n) p[i] = v;
}

__global__ void k_init_minmax(unsigned* mm) { mm[0] = 0xFFFFFFFFu; mm[1] = 0u; }

__device__ __forceinline__ unsigned fenc(float f) {
  unsigned u = __float_as_uint(f);
  return (u & 0x80000000u) ? ~u : (u | 0x80000000u);
}
__device__ __forceinline__ float fdec(unsigned u) {
  return __uint_as_float((u & 0x80000000u) ? (u & 0x7fffffffu) : ~u);
}

// ---------- degree ----------
__global__ void k_degree(const int* __restrict__ dst, float* __restrict__ deg) {
  int i = blockIdx.x * blockDim.x + threadIdx.x;
  if (i < N_EDGES) atomicAdd(&deg[dst[i]], 1.0f);
}
__global__ void k_deg_finish(float* __restrict__ deg_inv_inplace, float* __restrict__ dinv) {
  int i = blockIdx.x * blockDim.x + threadIdx.x;
  if (i < N_NODES) {
    float d = deg_inv_inplace[i];
    dinv[i] = rsqrtf(d);
    deg_inv_inplace[i] = 1.0f / d;   // becomes 1/deg
  }
}

// ---------- fp32 GEMM via V_WMMA_F32_16X16X4_F32:  C[M x Nout] = A[M x K] @ W[K x Nout]
// 4 waves per block, one 16x16 output tile per wave.
__global__ void k_gemm_f32_wmma(const float* __restrict__ A, const float* __restrict__ W,
                                float* __restrict__ C, int mtiles, int Kdim, int Nout) {
  int wave = threadIdx.x >> 5;
  int lane = threadIdx.x & 31;
  int tm = blockIdx.x * 4 + wave;
  if (tm >= mtiles) return;          // wave-uniform exit, EXEC stays full
  int tn = blockIdx.y;
  int r  = lane & 15;
  int hi = lane >> 4;
  const float* arow = A + (long)(tm * 16 + r) * Kdim + hi * 2; // A[row][k0+hi*2+{0,1}]
  const float* bcol = W + tn * 16 + r;                         // W[k][col]
  v8f acc = {};
  for (int k0 = 0; k0 < Kdim; k0 += 4) {
    v2f a, b;
    a.x = arow[k0];
    a.y = arow[k0 + 1];
    int kb = k0 + hi * 2;
    b.x = bcol[(long)kb * Nout];
    b.y = bcol[(long)(kb + 1) * Nout];
    acc = __builtin_amdgcn_wmma_f32_16x16x4_f32(false, a, false, b, (short)0, acc,
                                                false, false);
  }
  // C/D layout: vgpr v -> row = 8*hi + v, col = r
  float* crow = C + (long)(tm * 16 + hi * 8) * Nout + tn * 16 + r;
#pragma unroll
  for (int v = 0; v < 8; ++v) crow[(long)v * Nout] = acc[v];
}

// ---------- edge scatter:  H[dst] += dinv[src]*dinv[dst] * T[src]  (one block per edge)
__global__ void k_scatter(const int* __restrict__ src, const int* __restrict__ dst,
                          const float* __restrict__ T, const float* __restrict__ dinv,
                          float* __restrict__ H, int F) {
  int e = blockIdx.x;
  int f = threadIdx.x;
  int s = src[e], d = dst[e];
  float coef = dinv[s] * dinv[d];
  atomicAdd(&H[(long)d * F + f], coef * T[(long)s * F + f]);
}

// ---------- epilogue:  H = (H + T/deg + b), optional relu (in place)
__global__ void k_epilogue(float* __restrict__ H, const float* __restrict__ T,
                           const float* __restrict__ inv_deg, const float* __restrict__ b,
                           int F, int do_relu) {
  int n = blockIdx.x, f = threadIdx.x;
  long idx = (long)n * F + f;
  float v = H[idx] + T[idx] * inv_deg[n] + b[f];
  H[idx] = do_relu ? fmaxf(v, 0.0f) : v;
}

// ---------- f32 -> f16 ----------
__global__ void k_cvt_f16(const float* __restrict__ x, _Float16* __restrict__ y, long n) {
  long i = (long)blockIdx.x * blockDim.x + threadIdx.x;
  if (i < n) y[i] = (_Float16)x[i];
}

// ---------- WMMA f16 fragment loaders (row-major [node][32] halves) ----------
__device__ __forceinline__ v16h load_fragA(const _Float16* __restrict__ base, int lane) {
  // A 16x32 f16: lanes0-15 hold K=0..15 of row=lane; lanes16-31 K=16..31
  int r = lane & 15, hi = lane >> 4;
  const _Float16* p = base + (long)r * F3C;
  return *(const v16h*)(p + hi * 16);
}
__device__ __forceinline__ v16h load_fragB(const _Float16* __restrict__ base, int lane) {
  // B 32x16 f16 with B = af2^T: column n of B = row n of af2; K-range = hi*16..+15
  int r = lane & 15, hi = lane >> 4;
  return *(const v16h*)(base + (long)r * F3C + hi * 16);
}

// ---------- similarity pass 1: global min/max over af1 @ af2^T ----------
// One A fragment per wave, swept across TNW consecutive n-tiles (register reuse of A).
__global__ void k_sim_minmax(const _Float16* __restrict__ a1, const _Float16* __restrict__ a2,
                             unsigned* __restrict__ mm) {
  int wave = threadIdx.x >> 5;
  int lane = threadIdx.x & 31;
  int widx = blockIdx.x * 8 + wave;
  if (widx >= MT * NG) return;        // wave-uniform
  int tm  = widx / NG;
  int tn0 = (widx - tm * NG) * TNW;
  v16h a = load_fragA(a1 + (long)tm * 16 * F3C, lane);
  float mn = 3.4e38f, mx = -3.4e38f;
#pragma unroll
  for (int j = 0; j < TNW; ++j) {
    int tn = tn0 + j;
    if (tn >= MT) break;              // wave-uniform
    v16h b = load_fragB(a2 + (long)tn * 16 * F3C, lane);
    v8f c = {};
    c = __builtin_amdgcn_wmma_f32_16x16x32_f16(false, a, false, b, (short)0, c, false, false);
#pragma unroll
    for (int v = 0; v < 8; ++v) { mn = fminf(mn, c[v]); mx = fmaxf(mx, c[v]); }
  }
  for (int off = 16; off >= 1; off >>= 1) {
    mn = fminf(mn, __shfl_xor(mn, off, 32));
    mx = fmaxf(mx, __shfl_xor(mx, off, 32));
  }
  if (lane == 0) {
    atomicMin(&mm[0], fenc(mn));
    atomicMax(&mm[1], fenc(mx));
  }
}

// ---------- similarity pass 2: 16-bin histogram ----------
__global__ void k_sim_hist(const _Float16* __restrict__ a1, const _Float16* __restrict__ a2,
                           const unsigned* __restrict__ mm, float* __restrict__ hist) {
  __shared__ float lh[BINS];
  if (threadIdx.x < BINS) lh[threadIdx.x] = 0.0f;
  __syncthreads();
  int wave = threadIdx.x >> 5;
  int lane = threadIdx.x & 31;
  int widx = blockIdx.x * 8 + wave;
  if (widx < MT * NG) {
    int tm  = widx / NG;
    int tn0 = (widx - tm * NG) * TNW;
    v16h a = load_fragA(a1 + (long)tm * 16 * F3C, lane);
    float lo = fdec(mm[0]);
    float hi = fdec(mm[1]);
    float scale = (float)BINS / (hi - lo + 1e-12f);
#pragma unroll
    for (int j = 0; j < TNW; ++j) {
      int tn = tn0 + j;
      if (tn >= MT) break;            // wave-uniform
      v16h b = load_fragB(a2 + (long)tn * 16 * F3C, lane);
      v8f c = {};
      c = __builtin_amdgcn_wmma_f32_16x16x32_f16(false, a, false, b, (short)0, c, false, false);
#pragma unroll
      for (int v = 0; v < 8; ++v) {
        int bin = (int)((c[v] - lo) * scale);
        bin = min(max(bin, 0), BINS - 1);
        atomicAdd(&lh[bin], 1.0f);
      }
    }
  }
  __syncthreads();
  if (threadIdx.x < BINS) atomicAdd(&hist[threadIdx.x], lh[threadIdx.x]);
}

// ---------- attention ----------
__global__ void k_colmean(const float* __restrict__ af, float* __restrict__ cm) {
  int f = blockIdx.x;                 // 32 blocks
  float s = 0.0f;
  for (int n = threadIdx.x; n < N_NODES; n += 256) s += af[(long)n * F3C + f];
  __shared__ float red[256];
  red[threadIdx.x] = s;
  __syncthreads();
  for (int o = 128; o >= 1; o >>= 1) {
    if ((int)threadIdx.x < o) red[threadIdx.x] += red[threadIdx.x + o];
    __syncthreads();
  }
  if (threadIdx.x == 0) cm[f] = red[0] * (1.0f / (float)N_NODES);
}

__global__ void k_gc(const float* __restrict__ cm, const float* __restrict__ attw,
                     float* __restrict__ gc) {
  int j = threadIdx.x;                // 32 threads
  float a = 0.0f;
#pragma unroll
  for (int i = 0; i < F3C; ++i) a += cm[i] * attw[i * F3C + j];
  gc[j] = tanhf(a);
}

__global__ void k_att_pool(const float* __restrict__ af, const float* __restrict__ gc,
                           float* __restrict__ p) {
  __shared__ float sgc[F3C];
  __shared__ float sp[F3C];
  int t = threadIdx.x;
  if (t < F3C) { sgc[t] = gc[t]; sp[t] = 0.0f; }
  __syncthreads();
  int n = blockIdx.x * 256 + t;
  if (n < N_NODES) {
    const float* x = af + (long)n * F3C;
    float d = 0.0f;
#pragma unroll
    for (int f = 0; f < F3C; ++f) d += x[f] * sgc[f];
    float sg = 1.0f / (1.0f + __expf(-d));
#pragma unroll
    for (int f = 0; f < F3C; ++f) atomicAdd(&sp[f], sg * x[f]);
  }
  __syncthreads();
  if (t < F3C) atomicAdd(&p[t], sp[t]);
}

// ---------- tensor network ----------
// M[f][k] = tn_wb[k][f] + sum_f2 tn_w[f][f2][k]*e[f2];  c[k] = tn_bias[k] + sum_f tn_wb[k][32+f]*e[f]
__global__ void k_tn_prep(const float* __restrict__ tnw, const float* __restrict__ tnwb,
                          const float* __restrict__ tnb, const float* __restrict__ e,
                          float* __restrict__ M, float* __restrict__ c) {
  int tid = threadIdx.x;              // 512 threads
  int f = tid >> 4, k = tid & 15;
  float acc = tnwb[k * (2 * F3C) + f];
#pragma unroll
  for (int f2 = 0; f2 < F3C; ++f2) acc += tnw[((long)f * F3C + f2) * KT + k] * e[f2];
  M[f * KT + k] = acc;
  if (tid < KT) {
    float cc = tnb[tid];
#pragma unroll
    for (int ff = 0; ff < F3C; ++ff) cc += tnwb[tid * (2 * F3C) + F3C + ff] * e[ff];
    c[tid] = cc;
  }
}

__global__ void k_tn_s0(const float* __restrict__ p1, const float* __restrict__ M1,
                        const float* __restrict__ c1, float* __restrict__ s0) {
  int k = threadIdx.x;                // 16 threads
  float a = c1[k];
#pragma unroll
  for (int f = 0; f < F3C; ++f) a += p1[f] * M1[f * KT + k];
  s0[k] = fmaxf(a, 0.0f);
}

// sum over nodes of relu(X @ M + c) -> out[16] (caller divides by N)
__global__ void k_tn_mean(const float* __restrict__ X, const float* __restrict__ M,
                          const float* __restrict__ c, float* __restrict__ out) {
  __shared__ float sM[F3C * KT];
  __shared__ float sc[KT];
  __shared__ float part[KT];
  int t = threadIdx.x;
  for (int i = t; i < F3C * KT; i += 256) sM[i] = M[i];
  if (t < KT) { sc[t] = c[t]; part[t] = 0.0f; }
  __syncthreads();
  int n = blockIdx.x * 256 + t;
  if (n < N_NODES) {
    float x[F3C];
    const float* xp = X + (long)n * F3C;
#pragma unroll
    for (int f = 0; f < F3C; ++f) x[f] = xp[f];
#pragma unroll
    for (int k = 0; k < KT; ++k) {
      float a = sc[k];
#pragma unroll
      for (int f = 0; f < F3C; ++f) a += x[f] * sM[f * KT + k];
      atomicAdd(&part[k], fmaxf(a, 0.0f));
    }
  }
  __syncthreads();
  if (t < KT) atomicAdd(&out[t], part[t]);
}

// ---------- final tiny MLP ----------
__global__ void k_final(const float* __restrict__ s0, const float* __restrict__ s1sum,
                        const float* __restrict__ s2sum, const float* __restrict__ hist,
                        const float* __restrict__ fc1w, const float* __restrict__ fc1b,
                        const float* __restrict__ fc2w, const float* __restrict__ fc2b,
                        float* __restrict__ out) {
  __shared__ float sc[3 * KT + BINS];
  __shared__ float hid[16];
  __shared__ float hsum;
  int t = threadIdx.x;                // 64 threads
  if (t == 0) {
    float s = 0.0f;
    for (int i = 0; i < BINS; ++i) s += hist[i];
    hsum = s;
  }
  __syncthreads();
  const float invn = 1.0f / (float)N_NODES;
  if (t < 16)      sc[t] = s0[t];
  else if (t < 32) sc[t] = s1sum[t - 16] * invn;
  else if (t < 48) sc[t] = s2sum[t - 32] * invn;
  else             sc[t] = hist[t - 48] / hsum;
  __syncthreads();
  if (t < 16) {
    float a = fc1b[t];
    for (int i = 0; i < 64; ++i) a += sc[i] * fc1w[i * 16 + t];
    hid[t] = fmaxf(a, 0.0f);
  }
  __syncthreads();
  if (t == 0) {
    float o = fc2b[0];
    for (int j = 0; j < 16; ++j) o += hid[j] * fc2w[j];
    out[0] = 1.0f / (1.0f + __expf(-o));
  }
}

// =====================================================================
extern "C" void kernel_launch(void* const* d_in, const int* in_sizes, int n_in,
                              void* d_out, int out_size, void* d_ws, size_t ws_size,
                              hipStream_t stream) {
  const float* feat1 = (const float*)d_in[0];
  const float* feat2 = (const float*)d_in[1];
  const int*   ei1   = (const int*)d_in[2];
  const int*   ei2   = (const int*)d_in[3];
  const float* W1 = (const float*)d_in[4];   const float* b1 = (const float*)d_in[5];
  const float* W2 = (const float*)d_in[6];   const float* b2 = (const float*)d_in[7];
  const float* W3 = (const float*)d_in[8];   const float* b3 = (const float*)d_in[9];
  const float* attw = (const float*)d_in[10];
  const float* tnw  = (const float*)d_in[11];
  const float* tnwb = (const float*)d_in[12];
  const float* tnb  = (const float*)d_in[13];
  const float* fc1w = (const float*)d_in[14]; const float* fc1b = (const float*)d_in[15];
  const float* fc2w = (const float*)d_in[16]; const float* fc2b = (const float*)d_in[17];
  float* out = (float*)d_out;

  // ---- workspace carve ----
  float* ws = (float*)d_ws;
  float* BT    = ws;                                   // N*128
  float* B1    = BT + (size_t)N_NODES * 128;           // N*128
  float* B2    = B1 + (size_t)N_NODES * 128;           // N*128
  float* AF1   = B2 + (size_t)N_NODES * 128;           // N*32
  float* AF2   = AF1 + (size_t)N_NODES * F3C;          // N*32
  float* DINV1 = AF2 + (size_t)N_NODES * F3C;          // N
  float* INV1  = DINV1 + N_NODES;                      // N (deg then 1/deg)
  float* DINV2 = INV1 + N_NODES;                       // N
  float* INV2  = DINV2 + N_NODES;                      // N
  _Float16* AF1H = (_Float16*)(INV2 + N_NODES);        // N*32 halves
  _Float16* AF2H = AF1H + (size_t)N_NODES * F3C;       // N*32 halves
  float* sm = (float*)(AF2H + (size_t)N_NODES * F3C);
  float* CM = sm;            float* GC = CM + 32;
  float* P1 = GC + 32;       float* P2 = P1 + 32;
  float* M1 = P2 + 32;       float* C1 = M1 + 512;
  float* M2 = C1 + 16;       float* C2 = M2 + 512;
  float* S0 = C2 + 16;       float* S1 = S0 + 16;
  float* S2 = S1 + 16;       float* HIST = S2 + 16;
  unsigned* MM = (unsigned*)(HIST + 16);

  const int TB = 256;
  const int nblkN = (N_NODES + TB - 1) / TB;
  const int nblkE = (N_EDGES + TB - 1) / TB;

  // ---- per-graph GCN pass ----
  for (int g = 0; g < 2; ++g) {
    const float* feat = g ? feat2 : feat1;
    const int* src = (g ? ei2 : ei1);
    const int* dst = src + N_EDGES;
    float* DINV = g ? DINV2 : DINV1;
    float* INV  = g ? INV2  : INV1;
    float* AF   = g ? AF2   : AF1;

    // degrees (self-loop included): deg = 1 + count
    k_fill<<<nblkN, TB, 0, stream>>>(INV, 1.0f, N_NODES);
    k_degree<<<nblkE, TB, 0, stream>>>(dst, INV);
    k_deg_finish<<<nblkN, TB, 0, stream>>>(INV, DINV);

    // layer 1: [N,128]@[128,128] -> relu
    {
      dim3 grd((MT + 3) / 4, F1C / 16);
      k_gemm_f32_wmma<<<grd, 128, 0, stream>>>(feat, W1, BT, MT, F_IN, F1C);
      k_fill<<<(int)(((long)N_NODES * F1C + TB - 1) / TB), TB, 0, stream>>>(B1, 0.0f, (long)N_NODES * F1C);
      k_scatter<<<N_EDGES, F1C, 0, stream>>>(src, dst, BT, DINV, B1, F1C);
      k_epilogue<<<N_NODES, F1C, 0, stream>>>(B1, BT, INV, b1, F1C, 1);
    }
    // layer 2: [N,128]@[128,64] -> relu
    {
      dim3 grd((MT + 3) / 4, F2C / 16);
      k_gemm_f32_wmma<<<grd, 128, 0, stream>>>(B1, W2, BT, MT, F1C, F2C);
      k_fill<<<(int)(((long)N_NODES * F2C + TB - 1) / TB), TB, 0, stream>>>(B2, 0.0f, (long)N_NODES * F2C);
      k_scatter<<<N_EDGES, F2C, 0, stream>>>(src, dst, BT, DINV, B2, F2C);
      k_epilogue<<<N_NODES, F2C, 0, stream>>>(B2, BT, INV, b2, F2C, 1);
    }
    // layer 3: [N,64]@[64,32] -> linear
    {
      dim3 grd((MT + 3) / 4, F3C / 16);
      k_gemm_f32_wmma<<<grd, 128, 0, stream>>>(B2, W3, BT, MT, F2C, F3C);
      k_fill<<<(int)(((long)N_NODES * F3C + TB - 1) / TB), TB, 0, stream>>>(AF, 0.0f, (long)N_NODES * F3C);
      k_scatter<<<N_EDGES, F3C, 0, stream>>>(src, dst, BT, DINV, AF, F3C);
      k_epilogue<<<N_NODES, F3C, 0, stream>>>(AF, BT, INV, b3, F3C, 0);
    }
  }

  // ---- histogram of af1 @ af2^T via f16 WMMA, two streaming passes ----
  k_cvt_f16<<<(int)(((long)N_NODES * F3C + TB - 1) / TB), TB, 0, stream>>>(AF1, AF1H, (long)N_NODES * F3C);
  k_cvt_f16<<<(int)(((long)N_NODES * F3C + TB - 1) / TB), TB, 0, stream>>>(AF2, AF2H, (long)N_NODES * F3C);
  k_init_minmax<<<1, 1, 0, stream>>>(MM);
  const int simBlocks = (MT * NG + 7) / 8;   // 8 waves per block, TNW n-tiles per wave
  k_sim_minmax<<<simBlocks, 256, 0, stream>>>(AF1H, AF2H, MM);
  k_fill<<<1, BINS, 0, stream>>>(HIST, 0.0f, BINS);
  k_sim_hist<<<simBlocks, 256, 0, stream>>>(AF1H, AF2H, MM, HIST);

  // ---- attention pooling (gc = tanh(colmean(af) @ att_w)) ----
  k_colmean<<<F3C, 256, 0, stream>>>(AF1, CM);
  k_gc<<<1, F3C, 0, stream>>>(CM, attw, GC);
  k_fill<<<1, F3C, 0, stream>>>(P1, 0.0f, F3C);
  k_att_pool<<<nblkN, 256, 0, stream>>>(AF1, GC, P1);

  k_colmean<<<F3C, 256, 0, stream>>>(AF2, CM);
  k_gc<<<1, F3C, 0, stream>>>(CM, attw, GC);
  k_fill<<<1, F3C, 0, stream>>>(P2, 0.0f, F3C);
  k_att_pool<<<nblkN, 256, 0, stream>>>(AF2, GC, P2);

  // ---- tensor network (folded: relu(X@M + c)) ----
  k_tn_prep<<<1, 512, 0, stream>>>(tnw, tnwb, tnb, P2, M1, C1);   // e = p2
  k_tn_prep<<<1, 512, 0, stream>>>(tnw, tnwb, tnb, P1, M2, C2);   // e = p1
  k_tn_s0<<<1, KT, 0, stream>>>(P1, M1, C1, S0);
  k_fill<<<1, KT, 0, stream>>>(S1, 0.0f, KT);
  k_fill<<<1, KT, 0, stream>>>(S2, 0.0f, KT);
  k_tn_mean<<<nblkN, 256, 0, stream>>>(AF1, M1, C1, S1);
  k_tn_mean<<<nblkN, 256, 0, stream>>>(AF2, M2, C2, S2);

  // ---- final MLP ----
  k_final<<<1, 64, 0, stream>>>(S0, S1, S2, HIST, fc1w, fc1b, fc2w, fc2b, out);
}
